// DistortionAwareConv2D_30803505447537
// MI455X (gfx1250) — matrix operand order repaired and checked by reading, not compiled, not run on hardware
//
#include <hip/hip_runtime.h>

// ---------------------------------------------------------------------------
// Deformable 3x3 conv, NHWC, B=2 H=W=384 C=32 K=9 F=32, fp32 in/out.
// GEMM view: M = B*H*W pixels, Kdim = K*C = 288, N = F = 32.
// Per wave: one 16-pixel x 32-output tile via V_WMMA_F32_16X16X32_BF16,
// with hi/lo bf16 splitting of A (bilinear samples) and B (weights) for
// near-fp32 accuracy: A*B ~= Ah*Bh + Ah*Bl + Al*Bh (fp32 accumulate).
// Weights are pre-split and pre-laid-out as WMMA B-fragments in LDS.
// bf16 conversion now uses native __bf16 casts so clang can emit packed
// v_cvt instructions instead of bit-twiddled RNE (big VALU reduction).
// ---------------------------------------------------------------------------

typedef __attribute__((ext_vector_type(16))) __bf16 v16bf;
typedef __attribute__((ext_vector_type(8)))  float  v8f;

#define BATCH   2
#define HH      384
#define WW      384
#define CC      32
#define KTAPS   9
#define FF      32
#define TILE    16
#define TOTAL_PIX   (BATCH * HH * WW)
#define TOTAL_TILES (TOTAL_PIX / TILE)   // 18432; W%16==0 -> tile never crosses a row

__global__ __launch_bounds__(128) void deform_conv_wmma(
    const float* __restrict__ x,      // [B,H,W,C]
    const float* __restrict__ off,    // [1,H,W,K,2] (0 = x_off, 1 = y_off)
    const float* __restrict__ ker,    // [K,C,F]
    const float* __restrict__ bias,   // [F]
    float* __restrict__ out)          // [B,H,W,F]
{
    // B-fragment store: [k][nblk][hi|lo][lane] x 32 bytes (16 bf16) = 36 KB
    __shared__ uint4 Bsh[KTAPS * 2 * 2 * 32 * 2];

    const int tid = threadIdx.x;

    // ---- Build hi/lo bf16 WMMA B-fragments for the weights, once per block.
    // B (32x16, bf16) layout: lane = 16*half + n; element e = ker[k][e<8 ? 8*half+e : 16+8*half+e-8][n+16*nblk]
    for (int s = tid; s < KTAPS * 2 * 32; s += blockDim.x) {
        const int lane_s = s & 31;
        const int nblk_s = (s >> 5) & 1;
        const int k_s    = s >> 6;
        const int n_s    = lane_s & 15;
        const int hs_s   = lane_s >> 4;
        v16bf vh, vl;
#pragma unroll
        for (int e = 0; e < 16; ++e) {
            const int c = (e < 8) ? (8 * hs_s + e) : (16 + 8 * hs_s + (e - 8));
            const float v = ker[(k_s * CC + c) * FF + n_s + 16 * nblk_s];
            const __bf16 hv = (__bf16)v;
            vh[e] = hv;
            vl[e] = (__bf16)(v - (float)hv);
        }
        const int hiSlot = ((k_s * 2 + nblk_s) * 2 + 0) * 32 + lane_s;
        __builtin_memcpy(&Bsh[hiSlot * 2], &vh, 32);
        __builtin_memcpy(&Bsh[(hiSlot + 32) * 2], &vl, 32);
    }
    __syncthreads();

    const int lane = tid & 31;
    const int m    = lane & 15;   // pixel within tile; also N column for C/D
    const int hsel = lane >> 4;   // half-wave select

    const float bias0 = bias[m];
    const float bias1 = bias[m + 16];

    const int wavesTotal = gridDim.x * (blockDim.x >> 5);
    const int gwave = blockIdx.x * (blockDim.x >> 5) + (tid >> 5);

    for (int tile = gwave; tile < TOTAL_TILES; tile += wavesTotal) {
        const int p0 = tile * TILE;
        const int bb = p0 / (HH * WW);
        const int hw = p0 - bb * (HH * WW);
        const int h  = hw / WW;
        const int w0 = hw - h * WW;
        const int w  = w0 + m;

        v8f acc0 = {0.f,0.f,0.f,0.f,0.f,0.f,0.f,0.f};
        v8f acc1 = {0.f,0.f,0.f,0.f,0.f,0.f,0.f,0.f};

        const float* xb = x + (size_t)bb * ((size_t)HH * WW * CC);
        const int offBase = (h * WW + w) * KTAPS * 2;

        for (int k = 0; k < KTAPS; ++k) {
            const int di = k / 3, dj = k % 3;
            const float xoff = off[offBase + 2 * k];
            const float yoff = off[offBase + 2 * k + 1];

            // padded coords, clamp to [0, 385]
            float yf = fminf(fmaxf((float)(h + di) + yoff, 0.f), 385.f);
            float xf = fminf(fmaxf((float)(w + dj) + xoff, 0.f), 385.f);
            const float y0f = floorf(yf), x0f = floorf(xf);
            const int y0 = (int)y0f, x0 = (int)x0f;
            const int y1 = min(y0 + 1, 385), x1 = min(x0 + 1, 385);
            const float wy1 = (float)y1 - yf, wy0 = yf - y0f;
            const float wx1 = (float)x1 - xf, wx0 = xf - x0f;

            // bilinear sample of this lane's 16 channels:
            // chunk0 = c[8*hsel .. +8), chunk1 = c[16+8*hsel .. +8)
            float smp[16];
#pragma unroll
            for (int e = 0; e < 16; ++e) smp[e] = 0.f;

            auto corner = [&](int yi, int xi, float q) {
                if (yi >= 1 && yi <= HH && xi >= 1 && xi <= WW) {
                    const float* p = xb + ((size_t)(yi - 1) * WW + (xi - 1)) * CC;
                    const float4 v0 = *(const float4*)(p + hsel * 8);
                    const float4 v1 = *(const float4*)(p + hsel * 8 + 4);
                    const float4 v2 = *(const float4*)(p + 16 + hsel * 8);
                    const float4 v3 = *(const float4*)(p + 16 + hsel * 8 + 4);
                    smp[0]  += q * v0.x; smp[1]  += q * v0.y;
                    smp[2]  += q * v0.z; smp[3]  += q * v0.w;
                    smp[4]  += q * v1.x; smp[5]  += q * v1.y;
                    smp[6]  += q * v1.z; smp[7]  += q * v1.w;
                    smp[8]  += q * v2.x; smp[9]  += q * v2.y;
                    smp[10] += q * v2.z; smp[11] += q * v2.w;
                    smp[12] += q * v3.x; smp[13] += q * v3.y;
                    smp[14] += q * v3.z; smp[15] += q * v3.w;
                }
            };
            corner(y0, x0, wy1 * wx1);
            corner(y0, x1, wy1 * wx0);
            corner(y1, x0, wy0 * wx1);
            corner(y1, x1, wy0 * wx0);

            // split samples into hi/lo bf16 A-fragments (native converts)
            v16bf vah, valo;
#pragma unroll
            for (int e = 0; e < 16; ++e) {
                const __bf16 hv = (__bf16)smp[e];
                vah[e]  = hv;
                valo[e] = (__bf16)(smp[e] - (float)hv);
            }

#pragma unroll
            for (int nblk = 0; nblk < 2; ++nblk) {
                const int hiSlot = ((k * 2 + nblk) * 2) * 32 + lane;
                const v16bf bh = *reinterpret_cast<const v16bf*>(&Bsh[hiSlot * 2]);
                const v16bf bl = *reinterpret_cast<const v16bf*>(&Bsh[(hiSlot + 32) * 2]);
                v8f& acc = nblk ? acc1 : acc0;
                acc = __builtin_amdgcn_wmma_f32_16x16x32_bf16(
                          false, vah,  false, bh, (short)0, acc, false, false);
                acc = __builtin_amdgcn_wmma_f32_16x16x32_bf16(
                          false, vah,  false, bl, (short)0, acc, false, false);
                acc = __builtin_amdgcn_wmma_f32_16x16x32_bf16(
                          false, valo, false, bh, (short)0, acc, false, false);
            }
        }

        // C/D layout: VGPR r -> row M = r + 8*hsel, col N = lane&15
        const size_t orow = (size_t)(p0 + 8 * hsel) * FF + m;
#pragma unroll
        for (int r = 0; r < 8; ++r) {
            out[orow + (size_t)r * FF]      = acc0[r] + bias0;
            out[orow + (size_t)r * FF + 16] = acc1[r] + bias1;
        }
    }
}

extern "C" void kernel_launch(void* const* d_in, const int* in_sizes, int n_in,
                              void* d_out, int out_size, void* d_ws, size_t ws_size,
                              hipStream_t stream) {
    (void)in_sizes; (void)n_in; (void)out_size; (void)d_ws; (void)ws_size;
    const float* x    = (const float*)d_in[0];
    const float* off  = (const float*)d_in[1];
    const float* ker  = (const float*)d_in[2];
    const float* bias = (const float*)d_in[3];
    float* out = (float*)d_out;

    // 576 blocks x 4 waves = 2304 waves; 18432 tiles -> 8 tiles/wave,
    // amortizing the per-block LDS weight-fragment build.
    dim3 grid(576), block(128);
    hipLaunchKernelGGL(deform_conv_wmma, grid, block, 0, stream,
                       x, off, ker, bias, out);
}
